// AdaptiveTransformerModel_71236327571472
// MI455X (gfx1250) — matrix operand order repaired and checked by reading, not compiled, not run on hardware
//
#include <hip/hip_runtime.h>
#include <hip/hip_bf16.h>
#include <math.h>

typedef __bf16 bf16_t;
typedef __attribute__((ext_vector_type(16))) __bf16 v16bf;
typedef __attribute__((ext_vector_type(8)))  __bf16 v8bf;
typedef __attribute__((ext_vector_type(8)))  float  v8f;

#define LNUM 12
#define HNUM 16
#define EDIM 1024
#define DHD  64
#define FDIM 4096
#define VOC  50257
#define NB   2
#define TT   512
#define BTOT (NB * TT)
#define MID  (LNUM / 2)

// ---------- scalar bf16 helpers ----------
static __device__ __forceinline__ bf16_t f2bf(float f) {
  unsigned u = __builtin_bit_cast(unsigned, f);
  unsigned r = u + 0x7FFFu + ((u >> 16) & 1u);
  unsigned short h = (unsigned short)(r >> 16);
  return __builtin_bit_cast(bf16_t, h);
}

// ---------- block reductions (wave32) ----------
static __device__ __forceinline__ float waveSum(float v) {
#pragma unroll
  for (int o = 16; o > 0; o >>= 1) v += __shfl_down(v, o, 32);
  return v;
}
static __device__ __forceinline__ float waveMax(float v) {
#pragma unroll
  for (int o = 16; o > 0; o >>= 1) v = fmaxf(v, __shfl_down(v, o, 32));
  return v;
}
static __device__ __forceinline__ float blockSum(float v, float* red) {
  int lane = threadIdx.x & 31, wid = threadIdx.x >> 5;
  v = waveSum(v);
  if (lane == 0) red[wid] = v;
  __syncthreads();
  v = (threadIdx.x < 8) ? red[threadIdx.x] : 0.0f;
  if (wid == 0) v = waveSum(v);
  if (threadIdx.x == 0) red[0] = v;
  __syncthreads();
  return red[0];
}
static __device__ __forceinline__ float blockMax(float v, float* red) {
  int lane = threadIdx.x & 31, wid = threadIdx.x >> 5;
  v = waveMax(v);
  if (lane == 0) red[wid] = v;
  __syncthreads();
  v = (threadIdx.x < 8) ? red[threadIdx.x] : -1e30f;
  if (wid == 0) v = waveMax(v);
  if (threadIdx.x == 0) red[0] = v;
  __syncthreads();
  return red[0];
}

// ---------- embedding: x = wte[id] + wpe[t] ----------
__global__ void k_embed(const int* __restrict__ ids, const float* __restrict__ wte,
                        const float* __restrict__ wpe, float* __restrict__ x) {
  int row = blockIdx.x;                 // 0..BTOT-1
  int t = row & (TT - 1);
  long long id = ids[row];
  const float* src = wte + id * EDIM;
  const float* pp  = wpe + (long long)t * EDIM;
  float* dst = x + (long long)row * EDIM;
  for (int e = threadIdx.x; e < EDIM; e += 256) dst[e] = src[e] + pp[e];
}

// ---------- flat fp32 -> bf16 ----------
__global__ void k_cvt(const float* __restrict__ src, bf16_t* __restrict__ dst, long long n) {
  long long i = (long long)blockIdx.x * 256 + threadIdx.x;
  if (i < n) dst[i] = f2bf(src[i]);
}

// ---------- batched transpose + convert: dst[z][c][r] = src[z][r][c] ----------
__global__ void k_cvt_t(const float* __restrict__ src, bf16_t* __restrict__ dst,
                        int R, int C, long long total) {
  long long i = (long long)blockIdx.x * 256 + threadIdx.x;
  if (i >= total) return;
  long long rc = (long long)R * C;
  long long z = i / rc;
  long long rem = i - z * rc;
  int c = (int)(rem / R);
  int r = (int)(rem - (long long)c * R);
  dst[i] = f2bf(src[z * rc + (long long)r * C + c]);
}

// ---------- LayerNorm -> bf16 ----------
__global__ void k_ln(const float* __restrict__ x, const float* __restrict__ g,
                     const float* __restrict__ b, bf16_t* __restrict__ out) {
  __shared__ float red1[8], red2[8];
  int row = blockIdx.x;
  const float* px = x + (long long)row * EDIM;
  float s = 0.0f, sq = 0.0f;
  for (int e = threadIdx.x; e < EDIM; e += 256) { float v = px[e]; s += v; sq += v * v; }
  float mean = blockSum(s, red1) * (1.0f / EDIM);
  float var  = blockSum(sq, red2) * (1.0f / EDIM) - mean * mean;
  float rstd = rsqrtf(var + 1e-5f);
  bf16_t* po = out + (long long)row * EDIM;
  for (int e = threadIdx.x; e < EDIM; e += 256)
    po[e] = f2bf((px[e] - mean) * rstd * g[e] + b[e]);
}

// ---------- effective attention output bias: sum_h gate[h]*bo[h][e] ----------
__global__ void k_boeff(const float* __restrict__ bo, const float* __restrict__ gate,
                        float* __restrict__ out) {
  int e = blockIdx.x * 256 + threadIdx.x;
  if (e >= EDIM) return;
  float s = 0.0f;
#pragma unroll
  for (int h = 0; h < HNUM; ++h) s += gate[h] * bo[(long long)h * EDIM + e];
  out[e] = s;
}

// ---------- causal softmax over scores rows, write bf16 probs ----------
__global__ void k_softmax(const float* __restrict__ scores, bf16_t* __restrict__ probs) {
  __shared__ float red1[8], red2[8];
  long long row = blockIdx.x;           // 0 .. NB*HNUM*TT-1
  int t = (int)(row & (TT - 1));
  const float* ps = scores + row * TT;
  bf16_t* pp = probs + row * TT;
  float m = -1e30f;
  for (int s = threadIdx.x; s < TT; s += 256)
    m = fmaxf(m, (s <= t) ? ps[s] : -1e30f);
  m = blockMax(m, red1);
  float sum = 0.0f;
  for (int s = threadIdx.x; s < TT; s += 256)
    sum += (s <= t) ? __expf(ps[s] - m) : 0.0f;
  sum = blockSum(sum, red2);
  float inv = 1.0f / sum;
  for (int s = threadIdx.x; s < TT; s += 256)
    pp[s] = f2bf((s <= t) ? __expf(ps[s] - m) * inv : 0.0f);
}

// ---------- concat [bf16(x), enc] for skip fusion ----------
__global__ void k_concat(const float* __restrict__ x, const bf16_t* __restrict__ encl,
                         bf16_t* __restrict__ cat) {
  int row = blockIdx.x;
  const float* px = x + (long long)row * EDIM;
  const bf16_t* pe = encl + (long long)row * EDIM;
  bf16_t* pc = cat + (long long)row * 2 * EDIM;
  for (int e = threadIdx.x; e < EDIM; e += 256) {
    pc[e] = f2bf(px[e]);
    pc[EDIM + e] = pe[e];
  }
}

// ---------- WMMA GEMM: C = act(A[MxK] * B^T[NxK] * alpha + bias) (+resid) ----------
// Block: 128 threads = 4 waves (2x2), block tile 128x128, wave tile 64x64.
// Per 32-K step: 16 v_wmma vs 8 fragment loads -> 0.5 loads/WMMA.
// modes: 0 = standard [m*ldc+n] (fp32 and/or bf16 out, batched via sC)
//        1 = bf16 out at [b][h][t][d]   (m=(b,t) over BTOT, n=(h,d))
//        2 = bf16 out at [b][h][d][t]
//        3 = batched z=(b,h): bf16 out at [b][t][h][d], scaled by gate[h]
struct GemmArgs {
  const bf16_t* A;
  const bf16_t* B;
  float*  Cf;
  bf16_t* Cbf;
  const float* bias;
  const float* resid;
  const float* gatep;
  long long sA, sB, sC;
  int M, N, K;
  int lda, ldb, ldc;
  float alpha;
  int mode;
  int gelu;
};

__global__ __launch_bounds__(128) void k_gemm(GemmArgs g) {
  const int lane = threadIdx.x & 31;
  const int wave = threadIdx.x >> 5;    // 0..3
  const int wm = wave >> 1;             // 0..1
  const int wn = wave & 1;              // 0..1
  const int l16 = lane & 15;
  const int lh  = lane >> 4;            // 0/1
  const int z = blockIdx.z;

  const bf16_t* A  = g.A + (long long)z * g.sA;
  const bf16_t* Bp = g.B + (long long)z * g.sB;
  const int m0 = blockIdx.y * 128 + wm * 64;
  const int n0 = blockIdx.x * 128 + wn * 64;

  v8f acc[4][4];
  v8f zero = {0.f, 0.f, 0.f, 0.f, 0.f, 0.f, 0.f, 0.f};
#pragma unroll
  for (int mt = 0; mt < 4; ++mt)
#pragma unroll
    for (int nt = 0; nt < 4; ++nt) acc[mt][nt] = zero;

  // A fragment per-lane base: row m0+mt*16+l16, K offset lh*8 (layout 7.12.2)
  const bf16_t* arow[4];
#pragma unroll
  for (int mt = 0; mt < 4; ++mt)
    arow[mt] = A + (long long)(m0 + mt * 16 + l16) * g.lda + lh * 8;

  // B^T fragment per-lane base: row n (clamped), K offset lh*8
  const bf16_t* brow[4];
#pragma unroll
  for (int nt = 0; nt < 4; ++nt) {
    int n = n0 + nt * 16 + l16;
    if (n >= g.N) n = g.N - 1;          // clamp: stores masked below
    brow[nt] = Bp + (long long)n * g.ldb + lh * 8;
  }

  for (int kk = 0; kk < g.K; kk += 32) {
    if (kk + 32 < g.K) {                // gfx1250 global_prefetch_b8
      __builtin_prefetch(arow[0] + kk + 32, 0, 1);
      __builtin_prefetch(arow[2] + kk + 32, 0, 1);
      __builtin_prefetch(brow[0] + kk + 32, 0, 1);
      __builtin_prefetch(brow[2] + kk + 32, 0, 1);
    }
    v16bf a[4], b[4];
#pragma unroll
    for (int mt = 0; mt < 4; ++mt) {
      v8bf lo = *(const v8bf*)(arow[mt] + kk);        // K = kk + lh*8 + 0..7
      v8bf hi = *(const v8bf*)(arow[mt] + kk + 16);   // K = kk + 16 + lh*8 + 0..7
      a[mt] = __builtin_shufflevector(lo, hi, 0, 1, 2, 3, 4, 5, 6, 7,
                                      8, 9, 10, 11, 12, 13, 14, 15);
    }
#pragma unroll
    for (int nt = 0; nt < 4; ++nt) {
      v8bf lo = *(const v8bf*)(brow[nt] + kk);        // K = kk + lh*8 + 0..7
      v8bf hi = *(const v8bf*)(brow[nt] + kk + 16);   // K = kk + 16 + lh*8 + 0..7
      b[nt] = __builtin_shufflevector(lo, hi, 0, 1, 2, 3, 4, 5, 6, 7,
                                      8, 9, 10, 11, 12, 13, 14, 15);
    }
#pragma unroll
    for (int mt = 0; mt < 4; ++mt)
#pragma unroll
      for (int nt = 0; nt < 4; ++nt)
        acc[mt][nt] = __builtin_amdgcn_wmma_f32_16x16x32_bf16(
            false, a[mt], false, b[nt], (short)0, acc[mt][nt], false, false);
  }

  float* Cf = g.Cf;
  bf16_t* Cb = g.Cbf;
  if (g.mode == 0) {
    if (Cf) Cf += (long long)z * g.sC;
    if (Cb) Cb += (long long)z * g.sC;
  }
  int bz = 0, hz = 0;
  float gs = 1.0f;
  if (g.mode == 3) {
    bz = z / HNUM; hz = z - bz * HNUM;
    if (g.gatep) gs = g.gatep[hz];
  }

#pragma unroll
  for (int mt = 0; mt < 4; ++mt) {
#pragma unroll
    for (int nt = 0; nt < 4; ++nt) {
      int n = n0 + nt * 16 + l16;
      if (n >= g.N) continue;
      float bv = g.bias ? g.bias[n] : 0.0f;
#pragma unroll
      for (int r = 0; r < 8; ++r) {
        int m = m0 + mt * 16 + lh * 8 + r;   // C/D layout: lanes16-31 -> M+8
        float v = acc[mt][nt][r] * g.alpha + bv;
        if (g.gelu) v = 0.5f * v * (1.0f + erff(v * 0.70710678118f));
        if (g.resid) v += g.resid[(long long)m * g.ldc + n];
        if (g.mode == 0) {
          long long off = (long long)m * g.ldc + n;
          if (Cf) Cf[off] = v;
          if (Cb) Cb[off] = f2bf(v);
        } else if (g.mode == 1) {
          int b_ = m / TT, t_ = m - b_ * TT, h_ = n / DHD, d_ = n - h_ * DHD;
          Cb[(((long long)(b_ * HNUM + h_)) * TT + t_) * DHD + d_] = f2bf(v);
        } else if (g.mode == 2) {
          int b_ = m / TT, t_ = m - b_ * TT, h_ = n / DHD, d_ = n - h_ * DHD;
          Cb[(((long long)(b_ * HNUM + h_)) * DHD + d_) * TT + t_] = f2bf(v);
        } else {
          Cb[(((long long)(bz * TT + m)) * HNUM + hz) * DHD + n] = f2bf(v * gs);
        }
      }
    }
  }
}

// ---------- host-side launch helper ----------
static void run_gemm(hipStream_t stream, const bf16_t* A, const bf16_t* B,
                     float* Cf, bf16_t* Cbf, const float* bias, const float* resid,
                     const float* gatep, int M, int N, int K, int lda, int ldb, int ldc,
                     long long sA, long long sB, long long sC, float alpha,
                     int mode, int gelu, int batch) {
  GemmArgs g;
  g.A = A; g.B = B; g.Cf = Cf; g.Cbf = Cbf;
  g.bias = bias; g.resid = resid; g.gatep = gatep;
  g.sA = sA; g.sB = sB; g.sC = sC;
  g.M = M; g.N = N; g.K = K; g.lda = lda; g.ldb = ldb; g.ldc = ldc;
  g.alpha = alpha; g.mode = mode; g.gelu = gelu;
  dim3 grid((N + 127) / 128, (M + 127) / 128, batch);
  k_gemm<<<grid, dim3(128), 0, stream>>>(g);
}

static inline int cdiv_ll(long long n, int d) { return (int)((n + d - 1) / d); }

extern "C" void kernel_launch(void* const* d_in, const int* in_sizes, int n_in,
                              void* d_out, int out_size, void* d_ws, size_t ws_size,
                              hipStream_t stream) {
  (void)in_sizes; (void)n_in; (void)out_size; (void)ws_size;

  const int*   ids  = (const int*)d_in[0];
  const float* wte  = (const float*)d_in[1];
  const float* wpe  = (const float*)d_in[2];
  const float* Wq   = (const float*)d_in[3];
  const float* bq   = (const float*)d_in[4];
  const float* Wk   = (const float*)d_in[5];
  const float* bk   = (const float*)d_in[6];
  const float* Wv   = (const float*)d_in[7];
  const float* bv   = (const float*)d_in[8];
  const float* Wo   = (const float*)d_in[9];
  const float* bo   = (const float*)d_in[10];
  const float* gate = (const float*)d_in[11];
  const float* ln1g = (const float*)d_in[12];
  const float* ln1b = (const float*)d_in[13];
  const float* ln2g = (const float*)d_in[14];
  const float* ln2b = (const float*)d_in[15];
  const float* w1   = (const float*)d_in[16];
  const float* b1   = (const float*)d_in[17];
  const float* w2   = (const float*)d_in[18];
  const float* b2   = (const float*)d_in[19];
  const float* skw  = (const float*)d_in[20];
  const float* skb  = (const float*)d_in[21];
  const float* lnfg = (const float*)d_in[22];
  const float* lnfb = (const float*)d_in[23];

  // ---- workspace carve-out ----
  char* ws = (char*)d_ws;
  auto alloc = [&](size_t bytes) -> char* {
    char* p = ws; ws += (bytes + 255) & ~(size_t)255; return p;
  };
  bf16_t* wte_bf = (bf16_t*)alloc((size_t)VOC * EDIM * 2);
  bf16_t* wq_bt  = (bf16_t*)alloc((size_t)EDIM * EDIM * 2);
  bf16_t* wk_bt  = (bf16_t*)alloc((size_t)EDIM * EDIM * 2);
  bf16_t* wv_bt  = (bf16_t*)alloc((size_t)EDIM * EDIM * 2);
  bf16_t* wo_bt  = (bf16_t*)alloc((size_t)EDIM * EDIM * 2);
  bf16_t* w1_bt  = (bf16_t*)alloc((size_t)EDIM * FDIM * 2);
  bf16_t* w2_bt  = (bf16_t*)alloc((size_t)EDIM * FDIM * 2);
  bf16_t* sk_bt  = (bf16_t*)alloc((size_t)2 * EDIM * EDIM * 2);
  float*  x      = (float*)alloc((size_t)BTOT * EDIM * 4);
  bf16_t* hln    = (bf16_t*)alloc((size_t)BTOT * EDIM * 2);
  bf16_t* qbf    = (bf16_t*)alloc((size_t)NB * HNUM * TT * DHD * 2);
  bf16_t* kbf    = (bf16_t*)alloc((size_t)NB * HNUM * TT * DHD * 2);
  bf16_t* vtbf   = (bf16_t*)alloc((size_t)NB * HNUM * TT * DHD * 2);
  float*  scores = (float*)alloc((size_t)NB * HNUM * TT * TT * 4);
  bf16_t* probs  = (bf16_t*)alloc((size_t)NB * HNUM * TT * TT * 2);
  bf16_t* obf    = (bf16_t*)alloc((size_t)BTOT * EDIM * 2);
  bf16_t* fact   = (bf16_t*)alloc((size_t)BTOT * FDIM * 2);
  bf16_t* enc_bf = (bf16_t*)alloc((size_t)MID * BTOT * EDIM * 2);
  bf16_t* cat    = (bf16_t*)alloc((size_t)BTOT * 2 * EDIM * 2);
  float*  bo_eff = (float*)alloc((size_t)EDIM * 4);

  // ---- embeddings + LM-head weight conversion ----
  k_embed<<<BTOT, 256, 0, stream>>>(ids, wte, wpe, x);
  long long nwte = (long long)VOC * EDIM;
  k_cvt<<<cdiv_ll(nwte, 256), 256, 0, stream>>>(wte, wte_bf, nwte);

  const long long wEE = (long long)EDIM * EDIM;       // per-layer qkv/wo elements
  const long long wEF = (long long)EDIM * FDIM;
  const long long wSK = (long long)2 * EDIM * EDIM;

  for (int l = 0; l < LNUM; ++l) {
    // per-layer weight views
    const float* Wql = Wq + (long long)l * wEE;
    const float* Wkl = Wk + (long long)l * wEE;
    const float* Wvl = Wv + (long long)l * wEE;
    const float* Wol = Wo + (long long)l * wEE;
    const float* w1l = w1 + (long long)l * wEF;
    const float* w2l = w2 + (long long)l * wEF;

    // weight -> bf16 N x K layout
    k_cvt_t<<<cdiv_ll(wEE, 256), 256, 0, stream>>>(Wql, wq_bt, EDIM, DHD, wEE);  // per-head ExDH -> [h*DH+d][e]
    k_cvt_t<<<cdiv_ll(wEE, 256), 256, 0, stream>>>(Wkl, wk_bt, EDIM, DHD, wEE);
    k_cvt_t<<<cdiv_ll(wEE, 256), 256, 0, stream>>>(Wvl, wv_bt, EDIM, DHD, wEE);
    k_cvt_t<<<cdiv_ll(wEE, 256), 256, 0, stream>>>(Wol, wo_bt, HNUM * DHD, EDIM, wEE); // [e][h*DH+d]
    k_cvt_t<<<cdiv_ll(wEF, 256), 256, 0, stream>>>(w1l, w1_bt, EDIM, FDIM, wEF);       // [f][e]
    k_cvt_t<<<cdiv_ll(wEF, 256), 256, 0, stream>>>(w2l, w2_bt, FDIM, EDIM, wEF);       // [e][f]
    if (l >= MID)
      k_cvt_t<<<cdiv_ll(wSK, 256), 256, 0, stream>>>(skw + (long long)l * wSK, sk_bt,
                                                     2 * EDIM, EDIM, wSK);             // [e][2e]
    k_boeff<<<EDIM / 256, 256, 0, stream>>>(bo + (long long)l * HNUM * EDIM,
                                            gate + (long long)l * HNUM, bo_eff);

    // ---- attention ----
    k_ln<<<BTOT, 256, 0, stream>>>(x, ln1g + (long long)l * EDIM, ln1b + (long long)l * EDIM, hln);

    const float* bql = bq + (long long)l * HNUM * DHD;
    const float* bkl = bk + (long long)l * HNUM * DHD;
    const float* bvl = bv + (long long)l * HNUM * DHD;
    // Q,K -> [b][h][t][d] bf16 ; V -> [b][h][d][t] bf16
    run_gemm(stream, hln, wq_bt, nullptr, qbf,  bql, nullptr, nullptr,
             BTOT, EDIM, EDIM, EDIM, EDIM, EDIM, 0, 0, 0, 1.0f, 1, 0, 1);
    run_gemm(stream, hln, wk_bt, nullptr, kbf,  bkl, nullptr, nullptr,
             BTOT, EDIM, EDIM, EDIM, EDIM, EDIM, 0, 0, 0, 1.0f, 1, 0, 1);
    run_gemm(stream, hln, wv_bt, nullptr, vtbf, bvl, nullptr, nullptr,
             BTOT, EDIM, EDIM, EDIM, EDIM, EDIM, 0, 0, 0, 1.0f, 2, 0, 1);

    // scores = Q K^T / sqrt(DH), batched over (b,h)
    run_gemm(stream, qbf, kbf, scores, nullptr, nullptr, nullptr, nullptr,
             TT, TT, DHD, DHD, DHD, TT,
             (long long)TT * DHD, (long long)TT * DHD, (long long)TT * TT,
             0.125f, 0, 0, NB * HNUM);
    k_softmax<<<NB * HNUM * TT, 256, 0, stream>>>(scores, probs);

    // o = probs @ V (gate-scaled) -> [b][t][h][d] bf16
    run_gemm(stream, probs, vtbf, nullptr, obf, nullptr, nullptr,
             gate + (long long)l * HNUM,
             TT, DHD, TT, TT, TT, 0,
             (long long)TT * TT, (long long)DHD * TT, 0,
             1.0f, 3, 0, NB * HNUM);

    // attn projection + bo_eff + residual -> x
    run_gemm(stream, obf, wo_bt, x, nullptr, bo_eff, x, nullptr,
             BTOT, EDIM, HNUM * DHD, HNUM * DHD, HNUM * DHD, EDIM,
             0, 0, 0, 1.0f, 0, 0, 1);

    if (l < MID)
      k_cvt<<<cdiv_ll((long long)BTOT * EDIM, 256), 256, 0, stream>>>(
          x, enc_bf + (long long)l * BTOT * EDIM, (long long)BTOT * EDIM);

    // ---- MLP ----
    k_ln<<<BTOT, 256, 0, stream>>>(x, ln2g + (long long)l * EDIM, ln2b + (long long)l * EDIM, hln);
    run_gemm(stream, hln, w1_bt, nullptr, fact, b1 + (long long)l * FDIM, nullptr, nullptr,
             BTOT, FDIM, EDIM, EDIM, EDIM, FDIM, 0, 0, 0, 1.0f, 0, 1, 1);
    run_gemm(stream, fact, w2_bt, x, nullptr, b2 + (long long)l * EDIM, x, nullptr,
             BTOT, EDIM, FDIM, FDIM, FDIM, EDIM, 0, 0, 0, 1.0f, 0, 0, 1);

    // ---- decoder skip fusion ----
    if (l >= MID) {
      int el = LNUM - l - 1;            // always < MID here
      k_concat<<<BTOT, 256, 0, stream>>>(x, enc_bf + (long long)el * BTOT * EDIM, cat);
      run_gemm(stream, cat, sk_bt, x, nullptr, skb + (long long)l * EDIM, nullptr, nullptr,
               BTOT, EDIM, 2 * EDIM, 2 * EDIM, 2 * EDIM, EDIM, 0, 0, 0, 1.0f, 0, 0, 1);
    }
  }

  // ---- final LN + tied LM head ----
  k_ln<<<BTOT, 256, 0, stream>>>(x, lnfg, lnfb, hln);
  run_gemm(stream, hln, wte_bf, (float*)d_out, nullptr, nullptr, nullptr, nullptr,
           BTOT, VOC, EDIM, EDIM, EDIM, VOC, 0, 0, 0, 1.0f, 0, 0, 1);
}
